// PointCloudAE_49529562857561
// MI455X (gfx1250) — matrix operand order repaired and checked by reading, not compiled, not run on hardware
//
#include <hip/hip_runtime.h>
#include <hip/hip_bf16.h>

// ---------------------------------------------------------------------------
// CDNA5 (gfx1250) GCN point-cloud autoencoder forward.
// Dense GEMMs: v_wmma_f32_16x16x32_f16 (wave32), B panel async-staged to LDS
// via global_load_async_to_lds_b128 (+ s_wait_asynccnt), A streamed from
// global with clause'd b128 loads + global_prefetch_b8.
// Sparse aggregation: edge-parallel gather + global_atomic_add_f32 scatter.
// ---------------------------------------------------------------------------

#define N_NODES 100000
#define N_EDGES 1600000
#define HID     256
#define LAT     128
#define PSZ     2048
#define EPS_BN  1e-5f

typedef __attribute__((ext_vector_type(16))) _Float16 v16h;
typedef __attribute__((ext_vector_type(8)))  float    v8f;

// ---------------------------------------------------------------------------
// x [1,2,N] -> h0 [N,2] row-major
__global__ void k_transpose_x(const float* __restrict__ x,
                              float* __restrict__ h0, int N) {
    int i = blockIdx.x * blockDim.x + threadIdx.x;
    if (i >= N) return;
    h0[i * 2 + 0] = x[0 * N + i];
    h0[i * 2 + 1] = x[1 * N + i];
}

__global__ void k_deg_init(float* __restrict__ deg, int N) {
    int i = blockIdx.x * blockDim.x + threadIdx.x;
    if (i < N) deg[i] = 1.0f;
}

__global__ void k_deg_edges(const int* __restrict__ dst,
                            float* __restrict__ deg, int E) {
    int e = blockIdx.x * blockDim.x + threadIdx.x;
    if (e < E) atomicAdd(&deg[dst[e]], 1.0f);
}

__global__ void k_dinv(float* __restrict__ deg, int N) {
    int i = blockIdx.x * blockDim.x + threadIdx.x;
    if (i < N) deg[i] = rsqrtf(deg[i]);
}

// self-loop term initializes agg: agg[i][c] = h[i][c] * dinv[i]^2
__global__ void k_agg_self(const float* __restrict__ h,
                           const float* __restrict__ dinv,
                           float* __restrict__ agg, int N, int C) {
    long long idx = (long long)blockIdx.x * blockDim.x + threadIdx.x;
    long long tot = (long long)N * C;
    if (idx >= tot) return;
    int i = (int)(idx / C);
    float dv = dinv[i];
    agg[idx] = h[idx] * dv * dv;
}

// edge scatter: agg[dst] += h[src] * dinv[src]*dinv[dst]
__global__ void k_agg_edges(const float* __restrict__ h,
                            const int* __restrict__ src,
                            const int* __restrict__ dst,
                            const float* __restrict__ dinv,
                            float* __restrict__ agg, int E, int C) {
    int e = blockIdx.x * blockDim.x + threadIdx.x;
    if (e >= E) return;
    int s = src[e], d = dst[e];
    float nrm = dinv[s] * dinv[d];
    const float* hs = h + (size_t)s * C;
    float* ad = agg + (size_t)d * C;
    if ((C & 3) == 0) {
        for (int c = 0; c < C; c += 4) {
            float4 v = *(const float4*)(hs + c);
            atomicAdd(&ad[c + 0], v.x * nrm);
            atomicAdd(&ad[c + 1], v.y * nrm);
            atomicAdd(&ad[c + 2], v.z * nrm);
            atomicAdd(&ad[c + 3], v.w * nrm);
        }
    } else {
        for (int c = 0; c < C; c++)
            atomicAdd(&ad[c], hs[c] * nrm);
    }
}

// ---------------------------------------------------------------------------
// Layer 1 (K=2): h[n][o] = relu(agg[n][0]*W1[0][o] + agg[n][1]*W1[1][o] + b1[o])
__global__ void k_gcn1(const float* __restrict__ agg,
                       const float* __restrict__ W1, const float* __restrict__ b1,
                       float* __restrict__ h, int N) {
    int idx = blockIdx.x * blockDim.x + threadIdx.x;
    if (idx >= N * 64) return;
    int n = idx >> 6, o = idx & 63;
    float v = agg[n * 2 + 0] * W1[o] + agg[n * 2 + 1] * W1[64 + o] + b1[o];
    h[idx] = fmaxf(v, 0.0f);
}

// ---------------------------------------------------------------------------
// Weight prep: BT16[n*K + k] = (f16) Bp[k*bsk + n*bsn]   (row-major [Nc,K])
__global__ void k_prep_bt(const float* __restrict__ Bp, _Float16* __restrict__ BT,
                          int K, int Nc, int bsk, int bsn) {
    int idx = blockIdx.x * blockDim.x + threadIdx.x;
    if (idx >= K * Nc) return;
    int n = idx / K, k = idx - n * K;
    BT[idx] = (_Float16)Bp[(size_t)k * bsk + (size_t)n * bsn];
}

// ---------------------------------------------------------------------------
// CDNA5 async copy: LDS[lds_off .. +15] = MEM[gaddr .. +15], per lane.
// Tracked by ASYNCcnt.
__device__ __forceinline__ void async_b128_to_lds(unsigned lds_off, const void* gaddr) {
    asm volatile("global_load_async_to_lds_b128 %0, %1, off"
                 :: "v"(lds_off), "v"(gaddr) : "memory");
}
__device__ __forceinline__ void wait_asynccnt0() {
    asm volatile("s_wait_asynccnt 0x0" ::: "memory");
}

// ---------------------------------------------------------------------------
// WMMA GEMM: C[M,Nc] = relu?(A[M,K] x BT^T + bias)
//   A  : f32 row-major [M,K], K % 32 == 0
//   BT : f16 row-major [Nc,K]  (BT[n][k] = B(k,n))
//   Nc % 64 == 0, M % 16 == 0.
// Block = 256 threads (8 waves). blockIdx.y selects a 64-column group whose
// 64xK f16 panel is async-staged into LDS once; each wave then computes one
// 16x64 output strip (4 WMMA accumulators) for its own M-tile.
__global__ void __launch_bounds__(256)
k_gemm_wmma(const float* __restrict__ A,
            const _Float16* __restrict__ BT,
            const float* __restrict__ bias,
            float* __restrict__ C,
            int M, int K, int Nc, int relu) {
    __shared__ __align__(32) _Float16 Bs[64 * 256];   // up to 32 KB panel

    const int tid  = threadIdx.x;
    const int lane = tid & 31;
    const int wid  = tid >> 5;
    const int bg   = blockIdx.y;                      // 64-column group

    // ---- async stage B panel (64*K halves, contiguous) into LDS ----
    {
        const char* srcp = (const char*)(BT + (size_t)bg * 64 * K);
        const unsigned lds0 = (unsigned)(size_t)(&Bs[0]);
        const int bytes = 64 * K * 2;                 // 8/16/32 KB -> multiple of 4096
        for (int ofs = tid * 16; ofs < bytes; ofs += 256 * 16)
            async_b128_to_lds(lds0 + ofs, srcp + ofs);
        wait_asynccnt0();
    }
    __syncthreads();

    const int tilesM = M >> 4;
    const int tm = blockIdx.x * 8 + wid;              // one M-tile per wave
    if (tm >= tilesM) return;                         // wave-uniform exit

    const int hi   = (lane >> 4) & 1;                 // lane half
    const int lidx = lane & 15;
    const float* __restrict__ Arow = A + (size_t)(tm * 16 + lidx) * K;

    v8f acc[4] = {};
    for (int k0 = 0; k0 < K; k0 += 32) {
        if (k0 + 32 < K)
            __builtin_prefetch(Arow + k0 + 32, 0, 1);   // global_prefetch_b8

        // A fragment: slot i -> K=k0+16*hi+i (16 contiguous)
        const int ka = k0 + 16 * hi;
        float4 a0 = *(const float4*)(Arow + ka);
        float4 a1 = *(const float4*)(Arow + ka + 4);
        float4 a2 = *(const float4*)(Arow + ka + 8);
        float4 a3 = *(const float4*)(Arow + ka + 12);
        v16h a;
        a[0]  = (_Float16)a0.x; a[1]  = (_Float16)a0.y;
        a[2]  = (_Float16)a0.z; a[3]  = (_Float16)a0.w;
        a[4]  = (_Float16)a1.x; a[5]  = (_Float16)a1.y;
        a[6]  = (_Float16)a1.z; a[7]  = (_Float16)a1.w;
        a[8]  = (_Float16)a2.x; a[9]  = (_Float16)a2.y;
        a[10] = (_Float16)a2.z; a[11] = (_Float16)a2.w;
        a[12] = (_Float16)a3.x; a[13] = (_Float16)a3.y;
        a[14] = (_Float16)a3.z; a[15] = (_Float16)a3.w;

        // B fragments from LDS: slot j -> K = k0 + 16*hi + j (16 contiguous halves)
        const int kb = k0 + 16 * hi;
        v16h b0 = *(const v16h*)(&Bs[(lidx +  0) * K + kb]);
        v16h b1 = *(const v16h*)(&Bs[(lidx + 16) * K + kb]);
        v16h b2 = *(const v16h*)(&Bs[(lidx + 32) * K + kb]);
        v16h b3 = *(const v16h*)(&Bs[(lidx + 48) * K + kb]);

        acc[0] = __builtin_amdgcn_wmma_f32_16x16x32_f16(false, a, false, b0,
                                                        (short)0, acc[0], false, false);
        acc[1] = __builtin_amdgcn_wmma_f32_16x16x32_f16(false, a, false, b1,
                                                        (short)0, acc[1], false, false);
        acc[2] = __builtin_amdgcn_wmma_f32_16x16x32_f16(false, a, false, b2,
                                                        (short)0, acc[2], false, false);
        acc[3] = __builtin_amdgcn_wmma_f32_16x16x32_f16(false, a, false, b3,
                                                        (short)0, acc[3], false, false);
    }

    // C/D layout: VGPR r -> row tm*16 + r + 8*hi, col = bg*64 + 16*g + lidx
    const int rbase = tm * 16 + 8 * hi;
#pragma unroll
    for (int g = 0; g < 4; g++) {
        const int col = bg * 64 + 16 * g + lidx;
        const float bv = bias[col];
#pragma unroll
        for (int r = 0; r < 8; r++) {
            float v = acc[g][r] + bv;
            if (relu) v = fmaxf(v, 0.0f);
            C[(size_t)(rbase + r) * Nc + col] = v;
        }
    }
}

// ---------------------------------------------------------------------------
__global__ void k_bn_stats(const float* __restrict__ h,
                           float* __restrict__ mean, float* __restrict__ var,
                           int N, int C) {
    __shared__ float ssum[256];
    __shared__ float ssq[256];
    const int c = blockIdx.x;
    const int t = threadIdx.x;
    float s = 0.0f, q = 0.0f;
    for (int i = t; i < N; i += blockDim.x) {
        float v = h[(size_t)i * C + c];
        s += v; q += v * v;
    }
    ssum[t] = s; ssq[t] = q;
    __syncthreads();
    for (int off = blockDim.x >> 1; off > 0; off >>= 1) {
        if (t < off) { ssum[t] += ssum[t + off]; ssq[t] += ssq[t + off]; }
        __syncthreads();
    }
    if (t == 0) {
        float mu = ssum[0] / (float)N;
        mean[c] = mu;
        var[c]  = ssq[0] / (float)N - mu * mu;
    }
}

__global__ void k_bn_apply(float* __restrict__ h,
                           const float* __restrict__ mean,
                           const float* __restrict__ var,
                           const float* __restrict__ g,
                           const float* __restrict__ be,
                           int N, int C, int relu) {
    long long idx = (long long)blockIdx.x * blockDim.x + threadIdx.x;
    long long tot = (long long)N * C;
    if (idx >= tot) return;
    int c = (int)(idx % C);
    float v = (h[idx] - mean[c]) * rsqrtf(var[c] + EPS_BN) * g[c] + be[c];
    if (relu) v = fmaxf(v, 0.0f);
    h[idx] = v;
}

__global__ void k_maxpool(const float* __restrict__ h, float* __restrict__ z,
                          int N, int C) {
    __shared__ float smax[256];
    const int c = blockIdx.x;
    const int t = threadIdx.x;
    float m = -3.402823466e+38f;
    for (int i = t; i < N; i += blockDim.x)
        m = fmaxf(m, h[(size_t)i * C + c]);
    smax[t] = m;
    __syncthreads();
    for (int off = blockDim.x >> 1; off > 0; off >>= 1) {
        if (t < off) smax[t] = fmaxf(smax[t], smax[t + off]);
        __syncthreads();
    }
    if (t == 0) z[c] = smax[0];
}

// decoder MLP: z[128] -> 256 -> 256 -> 4096, single block of 256 threads
__global__ void k_decoder(const float* __restrict__ z,
                          const float* __restrict__ d1w, const float* __restrict__ d1b,
                          const float* __restrict__ d2w, const float* __restrict__ d2b,
                          const float* __restrict__ d3w, const float* __restrict__ d3b,
                          float* __restrict__ out) {
    __shared__ float zz[LAT];
    __shared__ float h1[HID];
    __shared__ float h2[HID];
    const int t = threadIdx.x;
    if (t < LAT) zz[t] = z[t];
    __syncthreads();
    {
        float a = d1b[t];
        const float* w = d1w + (size_t)t * LAT;
        for (int k = 0; k < LAT; k++) a += zz[k] * w[k];
        h1[t] = fmaxf(a, 0.0f);
    }
    __syncthreads();
    {
        float a = d2b[t];
        const float* w = d2w + (size_t)t * HID;
        for (int k = 0; k < HID; k++) a += h1[k] * w[k];
        h2[t] = fmaxf(a, 0.0f);
    }
    __syncthreads();
    for (int o = t; o < PSZ * 2; o += blockDim.x) {
        float a = d3b[o];
        const float* w = d3w + (size_t)o * HID;
        for (int k = 0; k < HID; k++) a += h2[k] * w[k];
        out[o] = a;
    }
}

// ---------------------------------------------------------------------------
static inline int ceil_div(int a, int b) { return (a + b - 1) / b; }

extern "C" void kernel_launch(void* const* d_in, const int* in_sizes, int n_in,
                              void* d_out, int out_size, void* d_ws, size_t ws_size,
                              hipStream_t stream) {
    (void)in_sizes; (void)n_in; (void)out_size; (void)ws_size;
    const int N = N_NODES, E = N_EDGES;

    const float* x   = (const float*)d_in[0];
    const int*   ei  = (const int*)d_in[1];
    const int*   src = ei;
    const int*   dst = ei + E;
    const float* W1  = (const float*)d_in[2];  const float* b1  = (const float*)d_in[3];
    const float* W2  = (const float*)d_in[4];  const float* b2  = (const float*)d_in[5];
    const float* W3  = (const float*)d_in[6];  const float* b3  = (const float*)d_in[7];
    const float* cw1 = (const float*)d_in[8];  const float* cb1 = (const float*)d_in[9];
    const float* g1  = (const float*)d_in[10]; const float* be1 = (const float*)d_in[11];
    const float* cw2 = (const float*)d_in[12]; const float* cb2 = (const float*)d_in[13];
    const float* g2  = (const float*)d_in[14]; const float* be2 = (const float*)d_in[15];
    const float* cw3 = (const float*)d_in[16]; const float* cb3 = (const float*)d_in[17];
    const float* g3  = (const float*)d_in[18]; const float* be3 = (const float*)d_in[19];
    const float* d1w = (const float*)d_in[20]; const float* d1b = (const float*)d_in[21];
    const float* d2w = (const float*)d_in[22]; const float* d2b = (const float*)d_in[23];
    const float* d3w = (const float*)d_in[24]; const float* d3b = (const float*)d_in[25];
    float* out = (float*)d_out;

    // workspace layout
    float* bufA = (float*)d_ws;                   // N * HID
    float* bufB = bufA + (size_t)N * HID;         // N * HID
    float* deg  = bufB + (size_t)N * HID;         // N   (becomes dinv)
    float* mean = deg + N;                        // HID
    float* var  = mean + HID;                     // HID
    float* zvec = var + HID;                      // LAT
    _Float16* wt16 = (_Float16*)(zvec + LAT);     // HID*HID halves (weight staging)

    const int T = 256;
    dim3 blk(T);

    // ---- normalization coefficients ----
    k_transpose_x<<<ceil_div(N, T), blk, 0, stream>>>(x, bufA, N);
    k_deg_init<<<ceil_div(N, T), blk, 0, stream>>>(deg, N);
    k_deg_edges<<<ceil_div(E, T), blk, 0, stream>>>(dst, deg, E);
    k_dinv<<<ceil_div(N, T), blk, 0, stream>>>(deg, N);

    auto agg = [&](const float* h, float* a, int C) {
        long long tot = (long long)N * C;
        k_agg_self<<<(unsigned)((tot + T - 1) / T), blk, 0, stream>>>(h, deg, a, N, C);
        k_agg_edges<<<ceil_div(E, T), blk, 0, stream>>>(h, src, dst, deg, a, E, C);
    };
    // prep weights into wt16 then run the WMMA GEMM
    auto gemm = [&](const float* A, const float* Bp, const float* bias, float* C,
                    int K, int Nc, int bsk, int bsn, int relu) {
        k_prep_bt<<<ceil_div(K * Nc, T), blk, 0, stream>>>(Bp, wt16, K, Nc, bsk, bsn);
        dim3 grid(ceil_div(N / 16, 8), Nc / 64);
        k_gemm_wmma<<<grid, blk, 0, stream>>>(A, wt16, bias, C, N, K, Nc, relu);
    };

    // ---- GCN stack (aggregate -> transform) ----
    agg(bufA, bufB, 2);
    k_gcn1<<<ceil_div(N * 64, T), blk, 0, stream>>>(bufB, W1, b1, bufA, N);
    agg(bufA, bufB, 64);
    gemm(bufB, W2, b2, bufA, /*K=*/64,  /*Nc=*/128, /*bsk=*/128, /*bsn=*/1, 1);
    agg(bufA, bufB, 128);
    gemm(bufB, W3, b3, bufA, /*K=*/128, /*Nc=*/256, /*bsk=*/256, /*bsn=*/1, 1);

    // ---- conv1d(k=1) + BN stack ----
    long long totH = (long long)N * HID;
    long long totL = (long long)N * LAT;

    gemm(bufA, cw1, cb1, bufB, HID, HID, /*bsk=*/1, /*bsn=*/HID, 0);
    k_bn_stats<<<HID, blk, 0, stream>>>(bufB, mean, var, N, HID);
    k_bn_apply<<<(unsigned)((totH + T - 1) / T), blk, 0, stream>>>(bufB, mean, var, g1, be1, N, HID, 1);

    gemm(bufB, cw2, cb2, bufA, HID, HID, /*bsk=*/1, /*bsn=*/HID, 0);
    k_bn_stats<<<HID, blk, 0, stream>>>(bufA, mean, var, N, HID);
    k_bn_apply<<<(unsigned)((totH + T - 1) / T), blk, 0, stream>>>(bufA, mean, var, g2, be2, N, HID, 1);

    gemm(bufA, cw3, cb3, bufB, HID, LAT, /*bsk=*/1, /*bsn=*/HID, 0);
    k_bn_stats<<<LAT, blk, 0, stream>>>(bufB, mean, var, N, LAT);
    k_bn_apply<<<(unsigned)((totL + T - 1) / T), blk, 0, stream>>>(bufB, mean, var, g3, be3, N, LAT, 0);

    // ---- global max pool + decoder ----
    k_maxpool<<<LAT, blk, 0, stream>>>(bufB, zvec, N, LAT);
    k_decoder<<<1, blk, 0, stream>>>(zvec, d1w, d1b, d2w, d2b, d3w, d3b, out);
}
